// ExpandFrame_60696477827174
// MI455X (gfx1250) — matrix operand order repaired and checked by reading, not compiled, not run on hardware
//
#include <hip/hip_runtime.h>
#include <hip/hip_bf16.h>
#include <math.h>

// Problem constants (from reference setup_inputs)
#define B_ 8
#define L_ 1024
#define D_ 512
#define BAND_R 40.0f   // |t-c| > ~36 -> fp32 exp underflows to exact 0; 40 is safe

typedef __attribute__((ext_vector_type(16))) _Float16 v16h;
typedef __attribute__((ext_vector_type(8)))  _Float16 v8h;
typedef __attribute__((ext_vector_type(8)))  float    v8f;

// ---------------------------------------------------------------------------
// Kernel 1: inclusive scan of durations -> segment centers c[b,l]
// c = cumsum(dur) - 0.5*round(dur). One 1024-thread block per batch row.
// ---------------------------------------------------------------------------
__global__ __launch_bounds__(1024)
void ef_centers_kernel(const float* __restrict__ dur, float* __restrict__ cen) {
    __shared__ float s[L_];
    const int b = blockIdx.x, tid = threadIdx.x;
    const float d = dur[b * L_ + tid];
    s[tid] = d;
    __syncthreads();
    #pragma unroll
    for (int off = 1; off < L_; off <<= 1) {
        float v = s[tid];
        float a = (tid >= off) ? s[tid - off] : 0.0f;
        __syncthreads();
        s[tid] = v + a;
        __syncthreads();
    }
    cen[b * L_ + tid] = s[tid] - 0.5f * rintf(d);  // rintf == round-half-even
}

// ---------------------------------------------------------------------------
// Kernel 2: hidden fp32 [B][L][D] -> f16 [B][D][L] (transposed), LDS-tiled so
// both global sides are coalesced. Lets the GEMM fetch WMMA A-operands as two
// aligned b128 loads per lane (K runs along l contiguously per d-row).
// ---------------------------------------------------------------------------
__global__ __launch_bounds__(256)
void ef_transpose_f16_kernel(const float* __restrict__ hid, _Float16* __restrict__ hT) {
    __shared__ float tile[32][33];
    const int b = blockIdx.z;
    const int d0 = blockIdx.x * 32, l0 = blockIdx.y * 32;
    const int tx = threadIdx.x, ty = threadIdx.y;  // block (32, 8)
    #pragma unroll
    for (int i = 0; i < 4; ++i)
        tile[ty + 8 * i][tx] = hid[((size_t)(b * L_ + l0 + ty + 8 * i)) * D_ + d0 + tx];
    __syncthreads();
    #pragma unroll
    for (int i = 0; i < 4; ++i)
        hT[((size_t)(b * D_ + d0 + ty + 8 * i)) * L_ + l0 + tx] =
            (_Float16)tile[tx][ty + 8 * i];
}

// ---------------------------------------------------------------------------
// Kernel 3: rden[b,t] = 1 / sum_l exp(temp*(t-c)^2). Banded via binary search
// on the monotone centers (terms outside the band are exact fp32 zeros).
// Empty band -> 1/0 = inf -> 0*inf = NaN, matching the reference's 0/0.
// ---------------------------------------------------------------------------
__global__ __launch_bounds__(256)
void ef_denom_kernel(const float* __restrict__ cen, const float* __restrict__ dur,
                     float* __restrict__ rden, int T) {
    const int b = blockIdx.y;
    const int t = blockIdx.x * 256 + threadIdx.x;
    if (t >= T) return;
    const float temp = -1.0f / (5.0f * sqrtf(dur[0]));
    const float* cb = cen + b * L_;
    const float tf = (float)t;
    const float loB = tf - BAND_R, hiB = tf + BAND_R;
    int lo = 0, hi = L_;
    while (lo < hi) { int m = (lo + hi) >> 1; if (cb[m] < loB) lo = m + 1; else hi = m; }
    int lo2 = lo, hi2 = L_;
    while (lo2 < hi2) { int m = (lo2 + hi2) >> 1; if (cb[m] <= hiB) lo2 = m + 1; else hi2 = m; }
    float s = 0.0f;
    for (int l = lo; l < lo2; ++l) {
        float dd = tf - cb[l];
        s += __expf(temp * dd * dd);
    }
    rden[b * T + t] = 1.0f / s;
}

// ---------------------------------------------------------------------------
// Kernel 4: banded WMMA GEMM. Grid (ceil(T/16), B), 256 threads = 8 waves.
// Wave w owns d in [64w, 64w+64): 4 C-tiles of 16x16 (v8f each).
// Per K-chunk of 32 phonemes:
//   B-operand (w-matrix) computed in-register: element i <-> K = 16*half + i,
//     so each lane reads 16 contiguous centers + 16 v_exp_f32.
//   A-operand: element i <-> K = 16*half + i, same order as B
//     (16 contiguous l per lane) => two aligned v8h global loads from hiddenT.
// All branches around the WMMAs are block-uniform (EXEC all-ones).
// ---------------------------------------------------------------------------
__global__ __launch_bounds__(256)
void ef_wmma_kernel(const _Float16* __restrict__ hT,   // [B][D][L] f16
                    const float* __restrict__ cen,     // [B][L]
                    const float* __restrict__ rden,    // [B][T]
                    const float* __restrict__ dur,
                    float* __restrict__ out, int T) {
    const int b    = blockIdx.y;
    const int t0   = blockIdx.x * 16;
    const int wave = threadIdx.x >> 5;
    const int lane = threadIdx.x & 31;
    const int n    = lane & 15;
    const int half = lane >> 4;
    const float temp = -1.0f / (5.0f * sqrtf(dur[0]));
    const float* cb = cen + b * L_;

    // Band of phonemes touching frames [t0, t0+15] (block-uniform bounds).
    const float loB = (float)t0 - BAND_R;
    const float hiB = (float)(t0 + 15) + BAND_R;
    int lo = 0, hi = L_;
    while (lo < hi) { int m = (lo + hi) >> 1; if (cb[m] < loB) lo = m + 1; else hi = m; }
    int lo2 = lo, hi2 = L_;
    while (lo2 < hi2) { int m = (lo2 + hi2) >> 1; if (cb[m] <= hiB) lo2 = m + 1; else hi2 = m; }
    const int lend = lo2;  // exclusive

    v8f acc[4] = {};
    const float tt   = (float)(t0 + n);
    const int  dbase = wave * 64;

    for (int kb = lo & ~31; kb < lend; kb += 32) {
        // ---- B operand: Gaussian weights, K = 16*half + i ----
        v16h bop;
        const int lb = kb + 16 * half;
        #pragma unroll
        for (int i = 0; i < 16; ++i) {
            float dd = tt - cb[lb + i];
            bop[i] = (_Float16)__expf(temp * dd * dd);
        }
        // ---- 4 A operands + WMMAs ----
        const int la = kb + 16 * half;
        #pragma unroll
        for (int jt = 0; jt < 4; ++jt) {
            const int d = dbase + jt * 16 + n;  // A: m = lane&15
            const _Float16* row = hT + ((size_t)(b * D_ + d)) * L_;
            v8h p0 = *(const v8h*)(row + la);        // K = 16*half + 0..7
            v8h p1 = *(const v8h*)(row + la + 8);    // K = 16*half + 8..15
            v16h aop;
            #pragma unroll
            for (int i = 0; i < 8; ++i) { aop[i] = p0[i]; aop[i + 8] = p1[i]; }
            acc[jt] = __builtin_amdgcn_wmma_f32_16x16x32_f16(
                false, aop, false, bop, (short)0, acc[jt], false, false);
        }
    }

    // Epilogue: scale column t by 1/denom, store out[b][d][t]. C/D layout:
    // element r holds M = r + 8*half, N = lane&15. Stores coalesce along t.
    const int t = t0 + n;
    const int tc = (t < T) ? t : (T - 1);
    const float rd = rden[b * T + tc];
    if (t < T) {
        #pragma unroll
        for (int jt = 0; jt < 4; ++jt) {
            #pragma unroll
            for (int r = 0; r < 8; ++r) {
                const int d = dbase + jt * 16 + r + 8 * half;
                out[((size_t)(b * D_ + d)) * T + t] = acc[jt][r] * rd;
            }
        }
    }
}

// ---------------------------------------------------------------------------
// Launch. T recovered from out_size (out is [B, D, T] fp32).
// Workspace: centers (32KB) | rden (B*T*4, 256B-aligned) | hiddenT f16 (8MB).
// ---------------------------------------------------------------------------
extern "C" void kernel_launch(void* const* d_in, const int* in_sizes, int n_in,
                              void* d_out, int out_size, void* d_ws, size_t ws_size,
                              hipStream_t stream) {
    const float* hid = (const float*)d_in[0];   // [B, L, D] fp32
    const float* dur = (const float*)d_in[1];   // [B, L]    fp32
    float* out = (float*)d_out;                 // [B, D, T] fp32
    const int T = out_size / (B_ * D_);

    char* ws = (char*)d_ws;
    float* cen  = (float*)ws;                                     // B*L*4 = 32KB
    float* rden = (float*)(ws + 32 * 1024);                       // B*T*4
    size_t rd_bytes = (((size_t)B_ * T * 4) + 255) & ~(size_t)255;
    _Float16* hT = (_Float16*)(ws + 32 * 1024 + rd_bytes);        // B*D*L*2 = 8MB

    ef_centers_kernel<<<B_, 1024, 0, stream>>>(dur, cen);
    ef_transpose_f16_kernel<<<dim3(D_ / 32, L_ / 32, B_), dim3(32, 8), 0, stream>>>(hid, hT);
    ef_denom_kernel<<<dim3((T + 255) / 256, B_), 256, 0, stream>>>(cen, dur, rden, T);
    ef_wmma_kernel<<<dim3((T + 15) / 16, B_), 256, 0, stream>>>(hT, cen, rden, dur, out, T);
}
